// MakeSphere_6528350290199
// MI455X (gfx1250) — matrix operand order repaired and checked by reading, not compiled, not run on hardware
//
#include <hip/hip_runtime.h>

#define NN 100000   // nodes
#define NF 200000   // faces
// H = 64 hidden channels; A-tile K = 128 = [x(64) | nb(64)]

typedef __attribute__((ext_vector_type(16))) __bf16 v16bf;
typedef __attribute__((ext_vector_type(8)))  __bf16 v8bf;
typedef __attribute__((ext_vector_type(8)))  float  v8f;
typedef __attribute__((ext_vector_type(4)))  float  v4f;

#if __has_builtin(__builtin_amdgcn_tensor_load_to_lds) && __has_builtin(__builtin_amdgcn_s_wait_tensorcnt)
#define USE_TDM 1
typedef __attribute__((ext_vector_type(4))) unsigned int tdm_v4u;
typedef __attribute__((ext_vector_type(8))) int          tdm_v8i;
typedef __attribute__((ext_vector_type(4))) int          tdm_v4i;
#endif

__device__ __forceinline__ unsigned short f2bf(float f) {
  union { float f; unsigned int u; } cv; cv.f = f;
  unsigned int u = cv.u;
  u += 0x7FFFu + ((u >> 16) & 1u);      // round-to-nearest-even
  return (unsigned short)(u >> 16);
}
__device__ __forceinline__ unsigned int pack2bf(float a, float b) {
  return (unsigned int)f2bf(a) | ((unsigned int)f2bf(b) << 16);
}

__global__ void k_zero(float* __restrict__ p, long long n) {
  long long i = (long long)blockIdx.x * blockDim.x + threadIdx.x;
  if (i < n) p[i] = 0.0f;
}

__global__ void k_counts(const int* __restrict__ faces, float* __restrict__ cnt, int F) {
  int f = blockIdx.x * blockDim.x + threadIdx.x;
  if (f >= F) return;
  unsafeAtomicAdd(&cnt[faces[3*f+0]], 1.0f);
  unsafeAtomicAdd(&cnt[faces[3*f+1]], 1.0f);
  unsafeAtomicAdd(&cnt[faces[3*f+2]], 1.0f);
}

// per-face local geometry: unit normal(3), edge lens(3), area(1); pad to 64ch
__global__ void k_extract_local(const float* __restrict__ pos, const int* __restrict__ faces,
                                float* __restrict__ out, int F) {
  int f = blockIdx.x * blockDim.x + threadIdx.x;
  if (f >= F) return;
  int i0 = faces[3*f+0], i1 = faces[3*f+1], i2 = faces[3*f+2];
  float ax = pos[3*i0+0], ay = pos[3*i0+1], az = pos[3*i0+2];
  float bx = pos[3*i1+0], by = pos[3*i1+1], bz = pos[3*i1+2];
  float cx = pos[3*i2+0], cy = pos[3*i2+1], cz = pos[3*i2+2];
  float e0x = bx-ax, e0y = by-ay, e0z = bz-az;
  float e1x = cx-ax, e1y = cy-ay, e1z = cz-az;
  float e2x = cx-bx, e2y = cy-by, e2z = cz-bz;
  float nx = e0y*e1z - e0z*e1y;
  float ny = e0z*e1x - e0x*e1z;
  float nz = e0x*e1y - e0y*e1x;
  float dn = sqrtf(nx*nx + ny*ny + nz*nz);
  float inv = 1.0f / (dn + 1e-8f);
  float* o = out + (long long)f * 64;
  o[0] = nx*inv; o[1] = ny*inv; o[2] = nz*inv;
  o[3] = sqrtf(e0x*e0x + e0y*e0y + e0z*e0z);
  o[4] = sqrtf(e1x*e1x + e1y*e1y + e1z*e1z);
  o[5] = sqrtf(e2x*e2x + e2y*e2y + e2z*e2z);
  o[6] = 0.5f * dn;
  #pragma unroll
  for (int c = 7; c < 64; ++c) o[c] = 0.0f;
}

// scatter-add face features to incident vertices (one thread per face*channel)
__global__ void k_scatter(const int* __restrict__ faces, const float* __restrict__ x,
                          float* __restrict__ node, int F) {
  long long t = (long long)blockIdx.x * blockDim.x + threadIdx.x;
  if (t >= (long long)F * 64) return;
  int c = (int)(t & 63);
  int f = (int)(t >> 6);
  float v = x[(long long)f * 64 + c];
  unsafeAtomicAdd(&node[(long long)faces[3*f+0]*64 + c], v);
  unsafeAtomicAdd(&node[(long long)faces[3*f+1]*64 + c], v);
  unsafeAtomicAdd(&node[(long long)faces[3*f+2]*64 + c], v);
}

__global__ void k_div(float* __restrict__ node, const float* __restrict__ cnt, int N) {
  long long t = (long long)blockIdx.x * blockDim.x + threadIdx.x;
  if (t >= (long long)N * 64) return;
  int i = (int)(t >> 6);
  node[t] = node[t] / fmaxf(cnt[i], 1.0f);
}

// once per layer: pack combined weights [ws;wn](128x64) -> bf16, n-major: wbt[n*128+k]
__global__ void k_wprep(const float* __restrict__ ws, const float* __restrict__ wn,
                        int wsK, int wnK, unsigned short* __restrict__ wbt) {
  int idx = blockIdx.x * blockDim.x + threadIdx.x;
  if (idx >= 64 * 128) return;
  int n = idx >> 7, k = idx & 127;
  float w = 0.0f;
  if (k < 64) { if (k < wsK) w = ws[k * 64 + n]; }
  else        { int kk = k - 64; if (kk < wnK) w = wn[kk * 64 + n]; }
  wbt[idx] = f2bf(w);
}

// out[F,64] = relu?( [x|nb](F,128) @ W(128,64) + b ), W prepacked bf16 n-major.
// block = 256 threads = 8 waves; each wave computes one 16-face x 64-out tile.
// Weights pulled global->LDS via one TDM tensor_load_to_lds per block (16KB from L2).
__global__ __launch_bounds__(256) void k_face_gemm(
    const int* __restrict__ faces, const float* __restrict__ x,
    const float* __restrict__ node, const unsigned short* __restrict__ wbt,
    const float* __restrict__ bias, float* __restrict__ out, int F, int relu)
{
  __shared__ __align__(16) unsigned short Bt[64 * 128];       // weights, n-major: Bt[n*128+k]
  __shared__ __align__(16) unsigned short Ash[8 * 16 * 128];  // per-wave A tiles, row-major

  int tid = threadIdx.x;
  int wave = tid >> 5, lane = tid & 31;

#if defined(USE_TDM)
  // One TDM descriptor per block: linear copy of 8192 bf16 (16KB) into LDS.
  if (tid == 0) {
    unsigned long long ga = (unsigned long long)(uintptr_t)wbt;
    unsigned int lds_off = (unsigned int)(uintptr_t)(void*)&Bt[0];  // low 32 bits = LDS byte offset
    tdm_v4u g0;
    g0[0] = 1u;                                              // count=1, is_restore=0, gather off
    g0[1] = lds_off;                                         // lds_addr [63:32]
    g0[2] = (unsigned int)(ga & 0xFFFFFFFFull);              // global_addr [95:64]
    g0[3] = (unsigned int)((ga >> 32) & 0x01FFFFFFull)       // global_addr [120:96]
            | (2u << 30);                                    // type=2 ("image") [127:126]
    const unsigned int NE = 64u * 128u;                      // 8192 2-byte elements
    tdm_v8i g1;
    g1[0] = (int)(1u << 16);                                 // wg_mask=0, data_size=1 (2B)
    g1[1] = (int)((NE & 0xFFFFu) << 16);                     // tensor_dim0[15:0] @ bits 63:48
    g1[2] = (int)(((NE >> 16) & 0xFFFFu) | (1u << 16));      // tensor_dim0[31:16]; tensor_dim1=1
    g1[3] = (int)((NE & 0xFFFFu) << 16);                     // tensor_dim1 hi=0; tile_dim0=NE
    g1[4] = 0;                                               // tile_dim1/2 unused (1D tile)
    g1[5] = (int)NE;                                         // tensor_dim0_stride lo
    g1[6] = 0;
    g1[7] = 0;
    tdm_v4i g2 = {0, 0, 0, 0};                               // dims 2..4 unused
    tdm_v4i g3 = {0, 0, 0, 0};
#if __clang_major__ >= 23
    tdm_v8i g4 = {0, 0, 0, 0, 0, 0, 0, 0};
    __builtin_amdgcn_tensor_load_to_lds(g0, g1, g2, g3, g4, 0);
#else
    __builtin_amdgcn_tensor_load_to_lds(g0, g1, g2, g3, 0);
#endif
    __builtin_amdgcn_s_wait_tensorcnt(0);
  }
#else
  // fallback: cooperative vectorized copy (16B per thread-iter)
  {
    const uint4* src = (const uint4*)wbt;
    uint4* dst = (uint4*)&Bt[0];
    for (int i = tid; i < (64 * 128) / 8; i += 256) dst[i] = src[i];
  }
#endif

  unsigned short* A = &Ash[wave * 16 * 128];
  long long faceBase = ((long long)blockIdx.x * 8 + wave) * 16;

  // stage A tile: 2 lanes per face row; half 0 = x, half 1 = nb (mean of 3 vertex means)
  {
    int row = lane >> 1, half = lane & 1;
    long long face = faceBase + row;
    unsigned int* d32 = (unsigned int*)&A[row * 128 + half * 64];
    if (face < F) {
      if (half == 0) {
        const v4f* xr = (const v4f*)&x[face * 64];
        #pragma unroll
        for (int q = 0; q < 16; ++q) {
          v4f v = xr[q];
          d32[q * 2 + 0] = pack2bf(v.x, v.y);
          d32[q * 2 + 1] = pack2bf(v.z, v.w);
        }
      } else {
        const int* fr = &faces[face * 3];
        const v4f* p0 = (const v4f*)&node[(long long)fr[0] * 64];
        const v4f* p1 = (const v4f*)&node[(long long)fr[1] * 64];
        const v4f* p2 = (const v4f*)&node[(long long)fr[2] * 64];
        #pragma unroll
        for (int q = 0; q < 16; ++q) {
          v4f v = (p0[q] + p1[q] + p2[q]) * (1.0f / 3.0f);
          d32[q * 2 + 0] = pack2bf(v.x, v.y);
          d32[q * 2 + 1] = pack2bf(v.z, v.w);
        }
      }
    } else {
      #pragma unroll
      for (int q = 0; q < 32; ++q) d32[q] = 0u;
    }
  }
  __syncthreads();

  int ln = lane & 15, lh = lane >> 4;

  v8f zero = {0.f, 0.f, 0.f, 0.f, 0.f, 0.f, 0.f, 0.f};
  v8f acc[4] = {zero, zero, zero, zero};

  #pragma unroll
  for (int kt = 0; kt < 4; ++kt) {
    int kb = kt * 32;
    // A fragment (16-bit A 16x32 layout): lane ln = row M, halves by lh
    v8bf alo = *(const v8bf*)&A[ln * 128 + kb + lh * 16];
    v8bf ahi = *(const v8bf*)&A[ln * 128 + kb + lh * 16 + 8];
    v16bf av;
    #pragma unroll
    for (int i = 0; i < 8; ++i) { av[i] = alo[i]; av[i + 8] = ahi[i]; }
    #pragma unroll
    for (int nt = 0; nt < 4; ++nt) {
      // B fragment (32x16): lane ln = column N, K = kb + lh*16 + 0..15 (contiguous in Bt)
      const unsigned short* bp = &Bt[(nt * 16 + ln) * 128 + kb + lh * 16];
      v8bf blo = *(const v8bf*)bp;
      v8bf bhi = *(const v8bf*)(bp + 8);
      v16bf bv;
      #pragma unroll
      for (int i = 0; i < 8; ++i) { bv[i] = blo[i]; bv[i + 8] = bhi[i]; }
      acc[nt] = __builtin_amdgcn_wmma_f32_16x16x32_bf16(
          false, av, false, bv, (short)0, acc[nt], false, false);
    }
  }

  // C/D layout: VGPR r -> row (lh*8 + r), column ln; add bias, relu, store
  #pragma unroll
  for (int nt = 0; nt < 4; ++nt) {
    float bv = bias[nt * 16 + ln];
    #pragma unroll
    for (int r = 0; r < 8; ++r) {
      long long face = faceBase + lh * 8 + r;
      if (face < F) {
        float v = acc[nt][r] + bv;
        if (relu) v = fmaxf(v, 0.0f);
        out[face * 64 + nt * 16 + ln] = v;
      }
    }
  }
}

// pos2 = pos + node @ wp(64x3) + bp(3)
__global__ void k_pos_update(const float* __restrict__ posIn, const float* __restrict__ node,
                             const float* __restrict__ wp, const float* __restrict__ bp,
                             float* __restrict__ posOut, int N) {
  int i = blockIdx.x * blockDim.x + threadIdx.x;
  if (i >= N) return;
  const float* nd = &node[(long long)i * 64];
  float a0 = bp[0], a1 = bp[1], a2 = bp[2];
  #pragma unroll 8
  for (int k = 0; k < 64; ++k) {
    float v = nd[k];
    a0 += v * wp[k * 3 + 0];
    a1 += v * wp[k * 3 + 1];
    a2 += v * wp[k * 3 + 2];
  }
  posOut[3 * i + 0] = posIn[3 * i + 0] + a0;
  posOut[3 * i + 1] = posIn[3 * i + 1] + a1;
  posOut[3 * i + 2] = posIn[3 * i + 2] + a2;
}

extern "C" void kernel_launch(void* const* d_in, const int* in_sizes, int n_in,
                              void* d_out, int out_size, void* d_ws, size_t ws_size,
                              hipStream_t stream) {
  const float* pos   = (const float*)d_in[0];
  const int*   faces = (const int*)d_in[1];
  const float* ws1  = (const float*)d_in[2],  *wn1  = (const float*)d_in[3],  *b1  = (const float*)d_in[4];
  const float* ws1a = (const float*)d_in[5],  *wn1a = (const float*)d_in[6],  *b1a = (const float*)d_in[7];
  const float* ws2  = (const float*)d_in[8],  *wn2  = (const float*)d_in[9],  *b2  = (const float*)d_in[10];
  const float* ws2a = (const float*)d_in[11], *wn2a = (const float*)d_in[12], *b2a = (const float*)d_in[13];
  const float* ws3  = (const float*)d_in[14], *wn3  = (const float*)d_in[15], *b3  = (const float*)d_in[16];
  const float* ws3a = (const float*)d_in[17], *wn3a = (const float*)d_in[18], *b3a = (const float*)d_in[19];
  const float* wp1 = (const float*)d_in[20], *bp1 = (const float*)d_in[21];
  const float* wf1 = (const float*)d_in[22], *bf1 = (const float*)d_in[23];
  const float* wp2 = (const float*)d_in[24], *bp2 = (const float*)d_in[25];
  const float* wf2 = (const float*)d_in[26], *bf2 = (const float*)d_in[27];
  const float* wp3 = (const float*)d_in[28], *bp3 = (const float*)d_in[29];
  float* outPos = (float*)d_out;

  // workspace: node[NN*64], cnt[NN], fA[NF*64], fB[NF*64], posA[NN*3], posB[NN*3], wbt[64*128 bf16]
  float* node = (float*)d_ws;
  float* cnt  = node + (size_t)NN * 64;
  float* fA   = cnt  + (size_t)NN;
  float* fB   = fA   + (size_t)NF * 64;
  float* posA = fB   + (size_t)NF * 64;
  float* posB = posA + (size_t)NN * 3;
  unsigned short* wbt = (unsigned short*)(posB + (size_t)NN * 3);

  const int TB = 256;
  auto nb1 = [](long long n) { return (int)((n + 255) / 256); };
  const int gemmBlocks = (NF + 127) / 128;   // 8 waves x 16 faces per block

  // vertex incidence counts (depend only on faces)
  k_zero<<<nb1(NN), TB, 0, stream>>>(cnt, (long long)NN);
  k_counts<<<nb1(NF), TB, 0, stream>>>(faces, cnt, NF);

  // initial per-face geometry features (padded to 64ch)
  k_extract_local<<<nb1(NF), TB, 0, stream>>>(pos, faces, fA, NF);

  auto node_mean = [&](const float* xin) {
    k_zero<<<nb1((long long)NN * 64), TB, 0, stream>>>(node, (long long)NN * 64);
    k_scatter<<<nb1((long long)NF * 64), TB, 0, stream>>>(faces, xin, node, NF);
    k_div<<<nb1((long long)NN * 64), TB, 0, stream>>>(node, cnt, NN);
  };
  auto gemm = [&](const float* xin, float* xout, const float* ws_, const float* wn_,
                  const float* b_, int wsK, int wnK, int relu) {
    k_wprep<<<32, TB, 0, stream>>>(ws_, wn_, wsK, wnK, wbt);
    k_face_gemm<<<gemmBlocks, TB, 0, stream>>>(faces, xin, node, wbt, b_, xout, NF, relu);
  };
  auto conv = [&](const float* xin, float* xout, const float* ws_, const float* wn_,
                  const float* b_, int wsK, int wnK) {
    node_mean(xin);
    gemm(xin, xout, ws_, wn_, b_, wsK, wnK, /*relu=*/1);
  };

  // block 1
  conv(fA, fB, ws1,  wn1,  b1,  7,  7);
  conv(fB, fA, ws1a, wn1a, b1a, 64, 64);
  node_mean(fA);
  k_pos_update<<<nb1(NN), TB, 0, stream>>>(pos, node, wp1, bp1, posA, NN);
  gemm(fA, fB, wf1, wf1, bf1, /*wsK=*/0, /*wnK=*/64, /*relu=*/0);
  // block 2
  conv(fB, fA, ws2,  wn2,  b2,  64, 64);
  conv(fA, fB, ws2a, wn2a, b2a, 64, 64);
  node_mean(fB);
  k_pos_update<<<nb1(NN), TB, 0, stream>>>(posA, node, wp2, bp2, posB, NN);
  gemm(fB, fA, wf2, wf2, bf2, 0, 64, 0);
  // block 3
  conv(fA, fB, ws3,  wn3,  b3,  64, 64);
  conv(fB, fA, ws3a, wn3a, b3a, 64, 64);
  node_mean(fA);
  k_pos_update<<<nb1(NN), TB, 0, stream>>>(posB, node, wp3, bp3, outPos, NN);
}
